// ManifoldHyperConnections_3075196584672
// MI455X (gfx1250) — compile-verified, matrix operand order checked by reading, not run on hardware
//
#include <hip/hip_runtime.h>
#include <hip/hip_bf16.h>
#include <math.h>

#define N_TOK 8192
#define DIM   2048
#define S     4
#define DFF   8192
#define SINKHORN_ITERS 10

typedef __attribute__((ext_vector_type(16))) __bf16 v16bf;
typedef __attribute__((ext_vector_type(8)))  __bf16 v8bf;
typedef __attribute__((ext_vector_type(8)))  float  v8f;
typedef int v4i __attribute__((vector_size(16)));   // type expected by async-LDS builtin

#if defined(__HIP_DEVICE_COMPILE__) && \
    __has_builtin(__builtin_amdgcn_global_load_async_to_lds_b128) && \
    __has_builtin(__builtin_amdgcn_s_wait_asynccnt)
#define HAS_ASYNC_LDS 1
#else
#define HAS_ASYNC_LDS 0
#endif

// ---------------- workspace layout (bytes) ----------------
static constexpr size_t OFF_WGATE = 0;
static constexpr size_t OFF_SRES  = OFF_WGATE + (size_t)24 * DIM * 4;
static constexpr size_t OFF_SPOST = OFF_SRES  + (size_t)N_TOK * 4;
static constexpr size_t OFF_BI    = OFF_SPOST + (size_t)N_TOK * 4;
static constexpr size_t OFF_W1T   = OFF_BI    + (size_t)N_TOK * DIM * 2;
static constexpr size_t OFF_W2T   = OFF_W1T   + (size_t)DIM * DFF * 2;
static constexpr size_t OFF_H     = OFF_W2T   + (size_t)DFF * DIM * 2;

// ================= weight prep =================
__global__ void phi_reduce_kernel(const float* __restrict__ pre_w,
                                  const float* __restrict__ post_w,
                                  const float* __restrict__ res_w,
                                  float* __restrict__ Wgate) {
  int idx = blockIdx.x * blockDim.x + threadIdx.x;   // 24 * DIM threads
  int r = idx / DIM, d = idx - r * DIM;
  const float* src;
  int row;
  if (r < S)          { src = pre_w;  row = r; }
  else if (r < 2 * S) { src = post_w; row = r - S; }
  else                { src = res_w;  row = r - 2 * S; }
  float s = 0.f;
#pragma unroll
  for (int j = 0; j < S; ++j)
    s += src[(size_t)row * (S * DIM) + (size_t)j * DIM + d];
  Wgate[idx] = s;
}

// fp32 [R][C] -> bf16 [C][R] (LDS-tiled transpose+convert)
__global__ __launch_bounds__(256)
void transpose_bf16_kernel(const float* __restrict__ in,
                           __hip_bfloat16* __restrict__ out,
                           int R, int C) {
  __shared__ float tile[32][33];
  int bx = blockIdx.x * 32;  // col base
  int by = blockIdx.y * 32;  // row base
  int tx = threadIdx.x;      // 0..31
  int ty = threadIdx.y;      // 0..7
#pragma unroll
  for (int i = ty; i < 32; i += 8)
    tile[i][tx] = in[(size_t)(by + i) * C + (bx + tx)];
  __syncthreads();
#pragma unroll
  for (int i = ty; i < 32; i += 8)
    out[(size_t)(bx + i) * R + (by + tx)] = __float2bfloat16(tile[tx][i]);
}

// ================= per-token gating =================
__global__ __launch_bounds__(256)
void gate_kernel(const float* __restrict__ x,
                 const float* __restrict__ Wgate,
                 const float* __restrict__ phi_pre_b,
                 const float* __restrict__ phi_post_b,
                 const float* __restrict__ phi_res_b,
                 const float* __restrict__ alpha_pre,
                 const float* __restrict__ alpha_post,
                 const float* __restrict__ alpha_res,
                 const float* __restrict__ b_pre,
                 const float* __restrict__ b_post,
                 const float* __restrict__ b_res,
                 __hip_bfloat16* __restrict__ block_input,
                 float* __restrict__ scale_res,
                 float* __restrict__ scale_post) {
  int t   = blockIdx.x;
  int tid = threadIdx.x;       // 0..255, 8 waves of 32
  int wid = tid >> 5, lane = tid & 31;
  const float* xt = x + (size_t)t * DIM;

  float xv[8];
#pragma unroll
  for (int i = 0; i < 8; ++i) xv[i] = xt[tid * 8 + i];

  float ss = 0.f;
#pragma unroll
  for (int i = 0; i < 8; ++i) ss += xv[i] * xv[i];

  float acc[24];
#pragma unroll
  for (int r = 0; r < 24; ++r) {
    const float* w = Wgate + (size_t)r * DIM + tid * 8;
    float a = 0.f;
#pragma unroll
    for (int i = 0; i < 8; ++i) a += xv[i] * w[i];
    acc[r] = a;
  }

  // wave32 shuffle reduction
#pragma unroll
  for (int off = 16; off > 0; off >>= 1) {
    ss += __shfl_down(ss, off, 32);
#pragma unroll
    for (int r = 0; r < 24; ++r) acc[r] += __shfl_down(acc[r], off, 32);
  }

  __shared__ float red[8][25];
  __shared__ float s_pre_bcast;
  if (lane == 0) {
    red[wid][24] = ss;
#pragma unroll
    for (int r = 0; r < 24; ++r) red[wid][r] = acc[r];
  }
  __syncthreads();

  if (tid == 0) {
    float dots[24], sstot = 0.f;
#pragma unroll
    for (int r = 0; r < 24; ++r) dots[r] = 0.f;
#pragma unroll
    for (int w = 0; w < 8; ++w) {
      sstot += red[w][24];
#pragma unroll
      for (int r = 0; r < 24; ++r) dots[r] += red[w][r];
    }
    float rn = rsqrtf(sstot * (1.0f / DIM) + 1.1920929e-7f);
    float ap = alpha_pre[0], aq = alpha_post[0], ar = alpha_res[0];

    float spre = 0.f, spost = 0.f;
#pragma unroll
    for (int s = 0; s < S; ++s) {
      float hp = ap * (dots[s] * rn + phi_pre_b[s]) + b_pre[s];
      spre += 1.0f / (1.0f + expf(-hp));
      float hq = aq * (dots[S + s] * rn + phi_post_b[s]) + b_post[s];
      spost += 2.0f / (1.0f + expf(-hq));
    }

    float M[16];
#pragma unroll
    for (int k = 0; k < 16; ++k)
      M[k] = expf(ar * (dots[8 + k] * rn + phi_res_b[k]) + b_res[k]);
    for (int it = 0; it < SINKHORN_ITERS; ++it) {
#pragma unroll
      for (int i = 0; i < 4; ++i) {
        float rs = M[4 * i] + M[4 * i + 1] + M[4 * i + 2] + M[4 * i + 3];
        float inv = 1.0f / rs;
        M[4 * i] *= inv; M[4 * i + 1] *= inv; M[4 * i + 2] *= inv; M[4 * i + 3] *= inv;
      }
#pragma unroll
      for (int j = 0; j < 4; ++j) {
        float cs = M[j] + M[4 + j] + M[8 + j] + M[12 + j];
        float inv = 1.0f / cs;
        M[j] *= inv; M[4 + j] *= inv; M[8 + j] *= inv; M[12 + j] *= inv;
      }
    }
    float sres = 0.f;
#pragma unroll
    for (int k = 0; k < 16; ++k) sres += M[k];

    scale_res[t]  = sres;
    scale_post[t] = spost;
    s_pre_bcast   = spre;
  }
  __syncthreads();

  float spre = s_pre_bcast;
  __hip_bfloat16* bo = block_input + (size_t)t * DIM + tid * 8;
#pragma unroll
  for (int i = 0; i < 8; ++i) bo[i] = __float2bfloat16(xv[i] * spre);
}

// ================= bf16 WMMA GEMM with async-LDS staging =================
// A: [rows][K] bf16 row-major.  Bm: [Ncols][K] bf16 (pre-transposed, K contiguous).
// Block: 128x128 tile, 8 waves (2x4), each wave 64x32 = 4x2 WMMA accumulators.
// LDS: double-buffered 128x32 tiles for A and B, row stride 40 elems (80 B:
// 16B-aligned chunks, bank-spread). Tiles staged with
// GLOBAL_LOAD_ASYNC_TO_LDS_B128 (ASYNCcnt), overlapping next-tile loads with
// current-tile WMMA; fragments read from LDS as 2x ds_load_b128.
// MODE 0: out bf16 = gelu_exact(acc + bias[col])
// MODE 1: out fp32 = x*scale_res[row] + (acc + bias[col])*scale_post[row]

#define LDS_STRIDE 40   // elements per tile row (32 data + 8 pad)

__device__ __forceinline__ void copy16_to_lds(const __bf16* g, __bf16* l) {
#if HAS_ASYNC_LDS
  __builtin_amdgcn_global_load_async_to_lds_b128((v4i*)g, (v4i*)l, 0, 0);
#else
  *(v8bf*)l = *(const v8bf*)g;
#endif
}

__device__ __forceinline__ void wait_stage_barrier() {
#if HAS_ASYNC_LDS
  __builtin_amdgcn_s_wait_asynccnt(0);
#endif
  __syncthreads();
}

__device__ __forceinline__ v16bf load_frag_lds(const __bf16* p) {
  union { v16bf v; v8bf h[2]; } u;
  u.h[0] = *(const v8bf*)p;
  u.h[1] = *(const v8bf*)(p + 8);
  return u.v;
}

template <int MODE>
__global__ __launch_bounds__(256)
void gemm_bf16_kernel(const __bf16* __restrict__ A,
                      const __bf16* __restrict__ Bm,
                      int K, int Ncols,
                      const float* __restrict__ bias,
                      const float* __restrict__ x,
                      const float* __restrict__ scale_res,
                      const float* __restrict__ scale_post,
                      void* __restrict__ outp) {
  __shared__ __bf16 ldsA[2][128 * LDS_STRIDE];
  __shared__ __bf16 ldsB[2][128 * LDS_STRIDE];

  const int bm = blockIdx.y * 128;
  const int bn = blockIdx.x * 128;
  const int tid = threadIdx.x, wid = tid >> 5, lane = tid & 31;
  const int wm = (wid >> 2) * 64;   // wave row offset within block tile
  const int wn = (wid & 3) * 32;    // wave col offset
  const int laneM = lane & 15;
  const int laneK = (lane >> 4) * 16;

  // Each tile (128 rows x 32 cols bf16) = 512 chunks of 16B; 2 chunks/thread.
  const int c0row = tid >> 2;             // chunk pass 0: rows 0..63
  const int c0col = (tid & 3) * 8;        // 16B chunk within row
  auto stage = [&](int buf, int k0) {
#pragma unroll
    for (int p = 0; p < 2; ++p) {
      const int row = c0row + p * 64;
      const size_t go = (size_t)row * K + k0 + c0col;
      const int    lo = row * LDS_STRIDE + c0col;
      copy16_to_lds(A  + (size_t)bm * K + go, &ldsA[buf][lo]);
      copy16_to_lds(Bm + (size_t)bn * K + go, &ldsB[buf][lo]);
    }
  };

  v8f acc[4][2] = {};
  const int nk = K / 32;

  stage(0, 0);
  for (int kt = 0; kt < nk; ++kt) {
    const int cur = kt & 1;
    wait_stage_barrier();                       // tile `cur` resident; prev reads done
    if (kt + 1 < nk) stage(cur ^ 1, (kt + 1) * 32);

    v16bf afrag[4], bfrag[2];
#pragma unroll
    for (int i = 0; i < 4; ++i)
      afrag[i] = load_frag_lds(&ldsA[cur][(wm + i * 16 + laneM) * LDS_STRIDE + laneK]);
#pragma unroll
    for (int j = 0; j < 2; ++j)
      bfrag[j] = load_frag_lds(&ldsB[cur][(wn + j * 16 + laneM) * LDS_STRIDE + laneK]);
#pragma unroll
    for (int i = 0; i < 4; ++i)
#pragma unroll
      for (int j = 0; j < 2; ++j)
        acc[i][j] = __builtin_amdgcn_wmma_f32_16x16x32_bf16(
            false, afrag[i], false, bfrag[j], (short)0, acc[i][j], false, false);
  }

  // C/D layout: VGPR r -> row = r + 8*(lane>=16); col = lane%16
  const int rowOff = (lane >> 4) * 8;
  const int col0   = lane & 15;
#pragma unroll
  for (int i = 0; i < 4; ++i) {
#pragma unroll
    for (int j = 0; j < 2; ++j) {
      const int col = bn + wn + j * 16 + col0;
      const float bcol = bias[col];
#pragma unroll
      for (int r = 0; r < 8; ++r) {
        const int row = bm + wm + i * 16 + rowOff + r;
        float v = acc[i][j][r] + bcol;
        if (MODE == 0) {
          v = 0.5f * v * (1.0f + erff(v * 0.70710678118654752f));  // exact GELU
          ((__hip_bfloat16*)outp)[(size_t)row * Ncols + col] = __float2bfloat16(v);
        } else {
          const size_t idx = (size_t)row * Ncols + col;
          ((float*)outp)[idx] = x[idx] * scale_res[row] + v * scale_post[row];
        }
      }
    }
  }
}

// ================= launch =================
extern "C" void kernel_launch(void* const* d_in, const int* in_sizes, int n_in,
                              void* d_out, int out_size, void* d_ws, size_t ws_size,
                              hipStream_t stream) {
  const float* x          = (const float*)d_in[0];
  const float* phi_pre_w  = (const float*)d_in[1];
  const float* phi_pre_b  = (const float*)d_in[2];
  const float* phi_post_w = (const float*)d_in[3];
  const float* phi_post_b = (const float*)d_in[4];
  const float* phi_res_w  = (const float*)d_in[5];
  const float* phi_res_b  = (const float*)d_in[6];
  const float* alpha_pre  = (const float*)d_in[7];
  const float* alpha_post = (const float*)d_in[8];
  const float* alpha_res  = (const float*)d_in[9];
  const float* b_pre      = (const float*)d_in[10];
  const float* b_post     = (const float*)d_in[11];
  const float* b_res      = (const float*)d_in[12];
  const float* W1         = (const float*)d_in[13];
  const float* b1         = (const float*)d_in[14];
  const float* W2         = (const float*)d_in[15];
  const float* b2         = (const float*)d_in[16];
  float* out = (float*)d_out;

  char* ws = (char*)d_ws;
  float*          Wgate = (float*)(ws + OFF_WGATE);
  float*          sres  = (float*)(ws + OFF_SRES);
  float*          spost = (float*)(ws + OFF_SPOST);
  __hip_bfloat16* BI    = (__hip_bfloat16*)(ws + OFF_BI);
  __hip_bfloat16* W1T   = (__hip_bfloat16*)(ws + OFF_W1T);
  __hip_bfloat16* W2T   = (__hip_bfloat16*)(ws + OFF_W2T);
  __hip_bfloat16* H     = (__hip_bfloat16*)(ws + OFF_H);

  // 1) fold S weight blocks -> Wgate[24][DIM]
  phi_reduce_kernel<<<(24 * DIM) / 256, 256, 0, stream>>>(phi_pre_w, phi_post_w,
                                                          phi_res_w, Wgate);
  // 2) W1 [DIM][DFF] -> W1T [DFF][DIM] bf16 ; W2 [DFF][DIM] -> W2T [DIM][DFF] bf16
  transpose_bf16_kernel<<<dim3(DFF / 32, DIM / 32), dim3(32, 8), 0, stream>>>(W1, W1T, DIM, DFF);
  transpose_bf16_kernel<<<dim3(DIM / 32, DFF / 32), dim3(32, 8), 0, stream>>>(W2, W2T, DFF, DIM);
  // 3) per-token gating
  gate_kernel<<<N_TOK, 256, 0, stream>>>(x, Wgate, phi_pre_b, phi_post_b, phi_res_b,
                                         alpha_pre, alpha_post, alpha_res,
                                         b_pre, b_post, b_res, BI, sres, spost);
  // 4) H = gelu(BI @ W1 + b1)  [N_TOK x DFF] bf16
  gemm_bf16_kernel<0><<<dim3(DFF / 128, N_TOK / 128), 256, 0, stream>>>(
      (const __bf16*)BI, (const __bf16*)W1T, DIM, DFF, b1,
      nullptr, nullptr, nullptr, (void*)H);
  // 5) out = x*scale_res + (H @ W2 + b2)*scale_post  [N_TOK x DIM] fp32
  gemm_bf16_kernel<1><<<dim3(DIM / 128, N_TOK / 128), 256, 0, stream>>>(
      (const __bf16*)H, (const __bf16*)W2T, DFF, DIM, b2,
      x, sres, spost, (void*)out);
}
